// ROIHead_73280732004498
// MI455X (gfx1250) — compile-verified, matrix operand order checked by reading, not run on hardware
//
#include <hip/hip_runtime.h>
#include <hip/hip_bf16.h>

typedef __attribute__((ext_vector_type(16))) _Float16 v16h;
typedef __attribute__((ext_vector_type(8)))  _Float16 v8h;
typedef __attribute__((ext_vector_type(8)))  float    v8f;
typedef __attribute__((ext_vector_type(4)))  float    v4f;

#define PP 7
#define KWIN 6
#define NEGV -1e30f
#define SCALE_ (1.0f/32.0f)
#define DW_CLIP_ 4.135166556742356f   /* log(1000/16) */
#define NMS_THR_ 0.3f
#define LOW_THR_ 0.05f
#define MIN_SIZE_ 16.0f
#define TOPK_ 100
#define M_CAND 1024

// GEMM tiling
#define BM 128
#define BN 128
#define BK 32
#define LDSK 40          /* padded row stride in halves: 80B -> conflict-free rows */

static __device__ __forceinline__ int imin(int a, int b) { return a < b ? a : b; }
static __device__ __forceinline__ int imax(int a, int b) { return a > b ? a : b; }

// ---------------------------------------------------------------------------
// Kernel 1: ROI max-pool.  One thread per (row, c, ph, pw) output element.
// Writes pooled activations as f16 (halves HBM traffic into FC6).
// ---------------------------------------------------------------------------
__global__ void roi_pool_kernel(const float* __restrict__ feat,   // B,C,H,W f32
                                const float* __restrict__ props,  // B,R,4 f32
                                _Float16* __restrict__ pooled,    // (B*R) x (C*49)
                                int Bn, int Cn, int Hn, int Wn, int Rn)
{
    size_t idx = (size_t)blockIdx.x * blockDim.x + threadIdx.x;
    size_t total = (size_t)Bn * Rn * Cn * PP * PP;
    if (idx >= total) return;
    int inDim = Cn * PP * PP;
    int col = (int)(idx % inDim);
    int row = (int)(idx / inDim);
    int c  = col / (PP * PP);
    int pp = col % (PP * PP);
    int ph = pp / PP, pw = pp % PP;
    int b = row / Rn, r = row % Rn;

    const float* pr = props + ((size_t)b * Rn + r) * 4;
    int x1 = (int)rintf(pr[0] * SCALE_);
    int y1 = (int)rintf(pr[1] * SCALE_);
    int x2 = (int)rintf(pr[2] * SCALE_);
    int y2 = (int)rintf(pr[3] * SCALE_);
    float rw = (float)imax(x2 - x1 + 1, 1);
    float rh = (float)imax(y2 - y1 + 1, 1);

    int hstart = imin(imax((int)floorf((float)ph * (rh / (float)PP)) + y1, 0), Hn);
    int hend   = imin(imax((int)ceilf(((float)ph + 1.0f) * (rh / (float)PP)) + y1, 0), Hn);
    int wstart = imin(imax((int)floorf((float)pw * (rw / (float)PP)) + x1, 0), Wn);
    int wend   = imin(imax((int)ceilf(((float)pw + 1.0f) * (rw / (float)PP)) + x1, 0), Wn);

    const float* f = feat + ((size_t)b * Cn + c) * (size_t)Hn * Wn;
    float m = NEGV;
    for (int kh = 0; kh < KWIN; ++kh) {
        int h = hstart + kh;
        if (h >= hend) break;
        for (int kw = 0; kw < KWIN; ++kw) {
            int w = wstart + kw;
            if (w >= wend) break;
            float v = f[h * Wn + w];
            m = m > v ? m : v;
        }
    }
    if (m <= NEGV * 0.5f) m = 0.0f;
    pooled[(size_t)row * inDim + col] = (_Float16)m;
}

// ---------------------------------------------------------------------------
// Kernel 2/3: double-buffered LDS-tiled GEMM C = relu(A * W^T + bias) via
// V_WMMA_F32_16X16X32_F16.  Block = 256 threads (8 waves), 128x128 C tile.
//   A tile (128x32 f16) : GLOBAL_LOAD_ASYNC_TO_LDS_B128 (ASYNCcnt), issued for
//                         tile k+1 before computing tile k.
//   B tile (128x32 f32) : global_load_b128 for k+1 issued pre-compute; cvt +
//                         ds_store_b128 post-compute; one barrier / iteration.
//   All 8 B fragments preloaded before the WMMA chain so the scheduler can use
//   partial dscnt waits (LDS returns in-order) and overlap ds_load with WMMA.
// ---------------------------------------------------------------------------
__global__ void gemm_f16_wmma(const _Float16* __restrict__ A,  // M x K f16
                              const float* __restrict__ W,     // N x K f32
                              const float* __restrict__ bias,  // N
                              _Float16* __restrict__ Cout,     // M x N f16
                              int M, int N, int Kd, int relu)
{
    __shared__ _Float16 As[2][BM * LDSK];
    __shared__ _Float16 Bs[2][BN * LDSK];

    const int tid  = threadIdx.x;
    const int lane = tid & 31;
    const int wv   = tid >> 5;          // 0..7
    const int m0 = blockIdx.x * BM;
    const int n0 = blockIdx.y * BN;

    // staging coordinates: 2 threads per row, 16 halves / 16 floats each
    const int srow = tid >> 1;          // 0..127
    const int skk  = (tid & 1) << 4;    // 0 or 16
    const int sOff = srow * LDSK + skk;

    const _Float16* gA = A + (size_t)(m0 + srow) * Kd + skk;
    const float*    gB = W + (size_t)(n0 + srow) * Kd + skk;
    const unsigned  ldsA0 = (unsigned)(size_t)&As[0][sOff];
    const unsigned  ldsA1 = (unsigned)(size_t)&As[1][sOff];

    // wave compute coordinates
    const int halfK = (lane >> 4) << 4;  // 0 or 16
    const int aOff  = (wv * 16 + (lane & 15)) * LDSK + halfK;

    auto loadBfrag = [&](int buf, int nt) -> v16h {
        const _Float16* bRow = &Bs[0][0] + buf * (BN * LDSK)
                               + (nt * 16 + (lane & 15)) * LDSK + halfK;
        v8h b0 = *(const v8h*)(bRow);
        v8h b1 = *(const v8h*)(bRow + 8);
        v16h r;
#pragma unroll
        for (int i = 0; i < 8; ++i) { r[i] = b0[i]; r[8 + i] = b1[i]; }
        return r;
    };
    auto cvtStoreB = [&](int buf, v4f f0, v4f f1, v4f f2, v4f f3) {
        v8h h0, h1;
#pragma unroll
        for (int i = 0; i < 4; ++i) {
            h0[i]     = (_Float16)f0[i];
            h0[4 + i] = (_Float16)f1[i];
            h1[i]     = (_Float16)f2[i];
            h1[4 + i] = (_Float16)f3[i];
        }
        // single shared-derived base + integer offset keeps addrspace(3)
        _Float16* p = &Bs[0][0] + buf * (BN * LDSK) + sOff;
        *(v8h*)(p)     = h0;
        *(v8h*)(p + 8) = h1;
    };

    v8f acc[8];
#pragma unroll
    for (int nt = 0; nt < 8; ++nt) acc[nt] = (v8f){};

    const int nsteps = Kd / BK;

    // ---- prologue: stage tile 0 into buffer 0
    {
        unsigned long long ga = (unsigned long long)(size_t)gA;
        asm volatile("global_load_async_to_lds_b128 %0, %1, off"
                     :: "v"(ldsA0), "v"(ga) : "memory");
        asm volatile("global_load_async_to_lds_b128 %0, %1, off offset:16"
                     :: "v"(ldsA0), "v"(ga) : "memory");
        const float* g = gB;
        cvtStoreB(0, *(const v4f*)(g), *(const v4f*)(g + 4),
                     *(const v4f*)(g + 8), *(const v4f*)(g + 12));
    }
    asm volatile("s_wait_asynccnt 0x0" ::: "memory");
    __syncthreads();

    for (int step = 0; step < nsteps; ++step) {
        const int cur = step & 1, nxt = cur ^ 1;
        const bool hasNext = (step + 1) < nsteps;
        const int k1 = (step + 1) * BK;

        // ---- issue next tile's loads before computing current tile
        v4f f0 = {}, f1 = {}, f2 = {}, f3 = {};
        if (hasNext) {
            unsigned la = nxt ? ldsA1 : ldsA0;
            unsigned long long ga = (unsigned long long)(size_t)(gA + k1);
            asm volatile("global_load_async_to_lds_b128 %0, %1, off"
                         :: "v"(la), "v"(ga) : "memory");
            asm volatile("global_load_async_to_lds_b128 %0, %1, off offset:16"
                         :: "v"(la), "v"(ga) : "memory");
            const float* g = gB + k1;
            f0 = *(const v4f*)(g);
            f1 = *(const v4f*)(g + 4);
            f2 = *(const v4f*)(g + 8);
            f3 = *(const v4f*)(g + 12);
            if (step + 2 < nsteps) __builtin_prefetch(gB + k1 + BK, 0, 3);
        }

        // ---- compute on current buffer
        const _Float16* aRow = &As[0][0] + cur * (BM * LDSK) + aOff;
        v8h a0 = *(const v8h*)(aRow);
        v8h a1 = *(const v8h*)(aRow + 8);
        v16h afrag;
#pragma unroll
        for (int i = 0; i < 8; ++i) { afrag[i] = a0[i]; afrag[8 + i] = a1[i]; }

        v16h bf[8];
#pragma unroll
        for (int nt = 0; nt < 8; ++nt) bf[nt] = loadBfrag(cur, nt);
#pragma unroll
        for (int nt = 0; nt < 8; ++nt) {
            acc[nt] = __builtin_amdgcn_wmma_f32_16x16x32_f16(
                false, afrag, false, bf[nt], (short)0, acc[nt], false, false);
        }

        // ---- finish staging next tile, single barrier per iteration
        if (hasNext) {
            cvtStoreB(nxt, f0, f1, f2, f3);
            asm volatile("s_wait_asynccnt 0x0" ::: "memory");
        }
        __syncthreads();
    }

    // ---- epilogue: bias + relu + f16 store, 16x128 per wave
    const int mbase = m0 + wv * 16 + ((lane >> 4) << 3);
#pragma unroll
    for (int nt = 0; nt < 8; ++nt) {
        const int cn = n0 + nt * 16 + (lane & 15);
        const float bv = bias[cn];
#pragma unroll
        for (int r = 0; r < 8; ++r) {
            float v = acc[nt][r] + bv;
            if (relu && v < 0.0f) v = 0.0f;
            Cout[(size_t)(mbase + r) * N + cn] = (_Float16)v;
        }
    }
}

// ---------------------------------------------------------------------------
// Kernel 4: tiny cls (5) + bbox (20) heads.  One thread per (row, output).
// ---------------------------------------------------------------------------
__global__ void heads_kernel(const _Float16* __restrict__ X,   // M x D f16
                             const float* __restrict__ cls_w, const float* __restrict__ cls_b,
                             const float* __restrict__ bbox_w, const float* __restrict__ bbox_b,
                             float* __restrict__ cls_out,      // M x 5
                             float* __restrict__ box_out,      // M x 20
                             int M, int Dd, int NC)
{
    int nout = NC + NC * 4;   // 25
    int t = blockIdx.x * blockDim.x + threadIdx.x;
    if (t >= M * nout) return;
    int row = t / nout, j = t % nout;
    const _Float16* x = X + (size_t)row * Dd;
    const float* w;
    float acc;
    if (j < NC) { w = cls_w  + (size_t)j * Dd;        acc = cls_b[j]; }
    else        { w = bbox_w + (size_t)(j - NC) * Dd; acc = bbox_b[j - NC]; }
    for (int k = 0; k < Dd; k += 4) {
        acc += (float)x[k]     * w[k];
        acc += (float)x[k + 1] * w[k + 1];
        acc += (float)x[k + 2] * w[k + 2];
        acc += (float)x[k + 3] * w[k + 3];
    }
    if (j < NC) cls_out[(size_t)row * NC + j] = acc;
    else        box_out[(size_t)row * (NC * 4) + (j - NC)] = acc;
}

// ---------------------------------------------------------------------------
// Kernel 5: per-image softmax + decode + clip + filter + sort + NMS + top-k.
// One 256-thread block per image; 1024 candidates in LDS; bitonic sort by
// score desc; serial-i suppression matching the reference fori_loop.
// ---------------------------------------------------------------------------
__global__ void filter_kernel(const float* __restrict__ props,  // B,R,4
                              const float* __restrict__ cls,    // (B*R) x 5
                              const float* __restrict__ boxt,   // (B*R) x 20
                              const int* __restrict__ img_sz,
                              float* __restrict__ out_boxes,    // B,100,4
                              float* __restrict__ out_labels,   // B,100
                              float* __restrict__ out_scores)   // B,100
{
    __shared__ float4 bx[M_CAND];
    __shared__ float  ssc[M_CAND];
    __shared__ int    sid[M_CAND];
    __shared__ int    keep[M_CAND];

    const int b = blockIdx.x;
    const int tid = threadIdx.x;
    const float img = (float)img_sz[0];

    // Phase 1: softmax, decode, clip, validity
    for (int r = tid; r < 256; r += blockDim.x) {
        int row = b * 256 + r;
        float lg[5];
        float mx = NEGV;
        for (int c = 0; c < 5; ++c) { lg[c] = cls[(size_t)row * 5 + c]; mx = fmaxf(mx, lg[c]); }
        float sum = 0.0f;
        for (int c = 0; c < 5; ++c) { lg[c] = expf(lg[c] - mx); sum += lg[c]; }
        float inv = 1.0f / sum;

        const float* pr = props + ((size_t)b * 256 + r) * 4;
        float w  = pr[2] - pr[0], h = pr[3] - pr[1];
        float cx = pr[0] + 0.5f * w, cy = pr[1] + 0.5f * h;

        for (int c = 1; c < 5; ++c) {
            int m = r * 4 + (c - 1);
            const float* t4 = boxt + (size_t)row * 20 + c * 4;
            float dx = t4[0], dy = t4[1];
            float dw = fminf(t4[2], DW_CLIP_), dh = fminf(t4[3], DW_CLIP_);
            float pcx = dx * w + cx, pcy = dy * h + cy;
            float pw = expf(dw) * w, ph = expf(dh) * h;
            float b0 = fminf(fmaxf(pcx - 0.5f * pw, 0.0f), img);
            float b1 = fminf(fmaxf(pcy - 0.5f * ph, 0.0f), img);
            float b2 = fminf(fmaxf(pcx + 0.5f * pw, 0.0f), img);
            float b3 = fminf(fmaxf(pcy + 0.5f * ph, 0.0f), img);
            float s = lg[c] * inv;
            bool valid = (s > LOW_THR_) && ((b2 - b0) >= MIN_SIZE_) && ((b3 - b1) >= MIN_SIZE_);
            bx[m] = make_float4(b0, b1, b2, b3);
            ssc[m] = valid ? s : -1.0f;
            sid[m] = m;
        }
    }
    __syncthreads();

    // Phase 2: bitonic sort (descending score), carry index
    for (int k = 2; k <= M_CAND; k <<= 1) {
        for (int j = k >> 1; j > 0; j >>= 1) {
            for (int i = tid; i < M_CAND; i += blockDim.x) {
                int ixj = i ^ j;
                if (ixj > i) {
                    bool up = ((i & k) == 0);   // keep larger at i -> descending
                    float sa = ssc[i], sb = ssc[ixj];
                    bool sw = up ? (sa < sb) : (sa > sb);
                    if (sw) {
                        ssc[i] = sb; ssc[ixj] = sa;
                        int t = sid[i]; sid[i] = sid[ixj]; sid[ixj] = t;
                    }
                }
            }
            __syncthreads();
        }
    }

    for (int p = tid; p < M_CAND; p += blockDim.x) keep[p] = (ssc[p] > 0.0f) ? 1 : 0;
    __syncthreads();

    // Phase 3: greedy NMS on class-offset boxes (matches the reference loop)
    const float offscale = img + 1000.0f;
    for (int i = 0; i < M_CAND - 1; ++i) {
        if (keep[i]) {
            int ia = sid[i];
            float offa = (float)(1 + (ia & 3)) * offscale;
            float4 A4 = bx[ia];
            float ax1 = A4.x + offa, ay1 = A4.y + offa;
            float ax2 = A4.z + offa, ay2 = A4.w + offa;
            float areaA = (ax2 - ax1) * (ay2 - ay1);
            for (int p = i + 1 + tid; p < M_CAND; p += blockDim.x) {
                if (!keep[p]) continue;
                int ib = sid[p];
                float offb = (float)(1 + (ib & 3)) * offscale;
                float4 B4 = bx[ib];
                float bx1 = B4.x + offb, by1 = B4.y + offb;
                float bx2 = B4.z + offb, by2 = B4.w + offb;
                float iw = fmaxf(fminf(ax2, bx2) - fmaxf(ax1, bx1), 0.0f);
                float ih = fmaxf(fminf(ay2, by2) - fmaxf(ay1, by1), 0.0f);
                float inter = iw * ih;
                float areaB = (bx2 - bx1) * (by2 - by1);
                float iou = inter / (areaA + areaB - inter + 1e-9f);
                if (iou > NMS_THR_) keep[p] = 0;
            }
        }
        __syncthreads();
    }

    // Phase 4: top-k emit (kept entries in sorted order, then -1 fillers)
    if (tid == 0) {
        int cnt = 0;
        for (int pass = 0; pass < 2 && cnt < TOPK_; ++pass) {
            for (int p = 0; p < M_CAND && cnt < TOPK_; ++p) {
                if ((pass == 0) != (keep[p] != 0)) continue;
                int ib = sid[p];
                float4 B4 = bx[ib];
                size_t o = (size_t)(b * TOPK_ + cnt);
                out_boxes[o * 4 + 0] = B4.x;
                out_boxes[o * 4 + 1] = B4.y;
                out_boxes[o * 4 + 2] = B4.z;
                out_boxes[o * 4 + 3] = B4.w;
                out_labels[o] = (float)(1 + (ib & 3));
                out_scores[o] = (pass == 0) ? ssc[p] : -1.0f;
                ++cnt;
            }
        }
    }
}

// ---------------------------------------------------------------------------
extern "C" void kernel_launch(void* const* d_in, const int* in_sizes, int n_in,
                              void* d_out, int out_size, void* d_ws, size_t ws_size,
                              hipStream_t stream) {
    const float* feat   = (const float*)d_in[0];
    const float* props  = (const float*)d_in[1];
    const int*   imgsz  = (const int*)d_in[2];
    const float* fc6_w  = (const float*)d_in[3];
    const float* fc6_b  = (const float*)d_in[4];
    const float* fc7_w  = (const float*)d_in[5];
    const float* fc7_b  = (const float*)d_in[6];
    const float* cls_w  = (const float*)d_in[7];
    const float* cls_b  = (const float*)d_in[8];
    const float* bbox_w = (const float*)d_in[9];
    const float* bbox_b = (const float*)d_in[10];
    float* out = (float*)d_out;

    const int Bn = 4, Cn = 256, Hn = 25, Wn = 25, Rn = 256, Dd = 1024, NC = 5;
    const int Mrows = Bn * Rn;            // 1024
    const int IN    = Cn * PP * PP;       // 12544

    // Workspace carve-out (~30 MB)
    char* ws = (char*)d_ws;
    size_t off = 0;
    auto carve = [&](size_t bytes) -> char* {
        char* p = ws + off;
        off = (off + bytes + 255) & ~(size_t)255;
        return p;
    };
    _Float16* pooled = (_Float16*)carve((size_t)Mrows * IN * sizeof(_Float16));
    _Float16* fc6a   = (_Float16*)carve((size_t)Mrows * Dd * sizeof(_Float16));
    _Float16* fc7a   = (_Float16*)carve((size_t)Mrows * Dd * sizeof(_Float16));
    float*    clsl   = (float*)carve((size_t)Mrows * NC * sizeof(float));
    float*    boxt   = (float*)carve((size_t)Mrows * NC * 4 * sizeof(float));

    // 1) ROI max-pool -> f16 activations
    {
        size_t total = (size_t)Mrows * IN;
        int threads = 256;
        int blocks = (int)((total + threads - 1) / threads);
        roi_pool_kernel<<<blocks, threads, 0, stream>>>(feat, props, pooled,
                                                        Bn, Cn, Hn, Wn, Rn);
    }

    // 2) FC6: 1024 x 12544 x 1024 (double-buffered LDS-tiled WMMA)
    // 3) FC7: 1024 x 1024  x 1024
    {
        dim3 block(256);                    // 8 waves / block
        dim3 grid(Mrows / BM, Dd / BN);     // (8, 8)
        gemm_f16_wmma<<<grid, block, 0, stream>>>(pooled, fc6_w, fc6_b, fc6a,
                                                  Mrows, Dd, IN, 1);
        gemm_f16_wmma<<<grid, block, 0, stream>>>(fc6a, fc7_w, fc7_b, fc7a,
                                                  Mrows, Dd, Dd, 1);
    }

    // 4) cls + bbox heads (N=25, trivial)
    {
        int total = Mrows * (NC + NC * 4);
        heads_kernel<<<(total + 127) / 128, 128, 0, stream>>>(
            fc7a, cls_w, cls_b, bbox_w, bbox_b, clsl, boxt, Mrows, Dd, NC);
    }

    // 5) softmax + decode + NMS + top-k; output = [boxes(4,100,4) | labels(4,100) | scores(4,100)]
    filter_kernel<<<Bn, 256, 0, stream>>>(props, clsl, boxt, imgsz,
                                          out, out + Bn * TOPK_ * 4,
                                          out + Bn * TOPK_ * 4 + Bn * TOPK_);
}
